// GroupedQueryAttention_42399917146871
// MI455X (gfx1250) — compile-verified, matrix operand order checked by reading, not run on hardware
//
#include <hip/hip_runtime.h>
#include <hip/hip_bf16.h>
#include <math.h>
#include <stdint.h>

#define NQ 24
#define NKV 4
#define HEAD_DIM 64
#define HIDDEN 1536
#define WINDOW 256
#define B_ 2
#define S_ 2048
#define ROWS (B_ * S_)                      // 4096
#define QKV_OUT ((NQ + 2 * NKV) * HEAD_DIM) // 2048

typedef __attribute__((ext_vector_type(8)))  __bf16 bf16x8;
typedef __attribute__((ext_vector_type(16))) __bf16 bf16x16;
typedef __attribute__((ext_vector_type(8)))  float  v8f;

// Pointer types matching the async-LDS builtin's signature:
// param0: AS1 (global / "__device__") int4-vector*, param1: AS3 (LDS) int4-vector*
typedef int int4v __attribute__((vector_size(16)));
typedef __attribute__((address_space(1))) int4v as1_int4;
typedef __attribute__((address_space(3))) int4v as3_int4;

#if defined(__HIP_DEVICE_COMPILE__) && __has_builtin(__builtin_amdgcn_global_load_async_to_lds_b128)
#define USE_ASYNC_LDS 1
#else
#define USE_ASYNC_LDS 0
#endif

__device__ __forceinline__ v8f vzero8() {
  v8f z = {0.f, 0.f, 0.f, 0.f, 0.f, 0.f, 0.f, 0.f};
  return z;
}

// Copy 16B global -> LDS. Async (gfx1250 GLOBAL_LOAD_ASYNC_TO_LDS_B128) when
// the builtin exists; plain LDS store fallback otherwise.
__device__ __forceinline__ void cp16(const __bf16* g, __bf16* l) {
#if USE_ASYNC_LDS
  __builtin_amdgcn_global_load_async_to_lds_b128(
      (as1_int4*)(uintptr_t)g, (as3_int4*)(uint32_t)(uintptr_t)l, 0, 0);
#else
  *reinterpret_cast<bf16x8*>(l) = *reinterpret_cast<const bf16x8*>(g);
#endif
}

__device__ __forceinline__ void await_async() {
#if USE_ASYNC_LDS
#if __has_builtin(__builtin_amdgcn_s_wait_asynccnt)
  __builtin_amdgcn_s_wait_asynccnt(0);
#else
  asm volatile("s_wait_asynccnt 0" ::: "memory");
#endif
#endif
}

// Load one 16-element bf16 fragment for lane-half `hi` from a global row ptr.
// Covers k-offsets {hi*8..hi*8+7} and {16+hi*8..+7} per the CDNA5 16-bit
// A/B VGPR layout (identical pattern for A rows and B columns of W[N][K]).
__device__ __forceinline__ bf16x16 load_frag(const __bf16* __restrict__ p, int hi) {
  bf16x8 lo = *reinterpret_cast<const bf16x8*>(p + hi * 8);
  bf16x8 hh = *reinterpret_cast<const bf16x8*>(p + 16 + hi * 8);
  return __builtin_shufflevector(lo, hh, 0, 1, 2, 3, 4, 5, 6, 7,
                                 8, 9, 10, 11, 12, 13, 14, 15);
}

// Same fragment read from a swizzled LDS tile [rows][32] (chunk c of row r is
// stored at chunk position c ^ ((r>>2)&3) to spread bank-quads).
__device__ __forceinline__ bf16x16 lds_frag(const __bf16* tile, int row, int hi) {
  int sw = (row >> 2) & 3;
  const bf16x8* p0 =
      reinterpret_cast<const bf16x8*>(tile + row * 32 + ((hi ^ sw) << 3));
  const bf16x8* p1 =
      reinterpret_cast<const bf16x8*>(tile + row * 32 + (((2 + hi) ^ sw) << 3));
  return __builtin_shufflevector(*p0, *p1, 0, 1, 2, 3, 4, 5, 6, 7,
                                 8, 9, 10, 11, 12, 13, 14, 15);
}

__device__ __forceinline__ v8f wmma_bf16f32(bf16x16 a, bf16x16 b, v8f c) {
  return __builtin_amdgcn_wmma_f32_16x16x32_bf16(false, a, false, b,
                                                 (short)0, c, false, false);
}

// ---------------- RMSNorm -> bf16 ----------------
__global__ void rmsnorm_kernel(const float* __restrict__ x,
                               const float* __restrict__ scale,
                               __bf16* __restrict__ xn) {
  __shared__ float red[256];
  int row = blockIdx.x;
  const float* xr = x + (size_t)row * HIDDEN;
  float s = 0.f;
  for (int i = threadIdx.x; i < HIDDEN; i += 256) {
    float v = xr[i];
    s += v * v;
  }
  red[threadIdx.x] = s;
  __syncthreads();
  for (int off = 128; off > 0; off >>= 1) {
    if ((int)threadIdx.x < off) red[threadIdx.x] += red[threadIdx.x + off];
    __syncthreads();
  }
  float inv = rsqrtf(red[0] * (1.0f / HIDDEN) + 1e-6f);
  __bf16* o = xn + (size_t)row * HIDDEN;
  for (int i = threadIdx.x; i < HIDDEN; i += 256)
    o[i] = (__bf16)(xr[i] * inv * scale[i]);
}

// ---------------- f32 -> bf16 convert ----------------
__global__ void cvt_bf16_kernel(const float* __restrict__ src,
                                __bf16* __restrict__ dst, int n) {
  for (int i = blockIdx.x * blockDim.x + threadIdx.x; i < n;
       i += gridDim.x * blockDim.x)
    dst[i] = (__bf16)src[i];
}

// ---- GEMM: C[M][N] = A[M][K] @ Bw[N][K]^T (+resid), LDS double-buffered ----
// Block = 8 waves, tile 128x128 (BK=32). Wave (wm,wn) = 32x64 sub-tile.
// A and B tiles staged through LDS with async global->LDS loads.
__global__ __launch_bounds__(256) void gemm_bf16_lds_kernel(
    const __bf16* __restrict__ A, const __bf16* __restrict__ Bw,
    const float* __restrict__ resid, float* __restrict__ C,
    int M, int N, int K) {
  __shared__ alignas(16) __bf16 sA[2][128 * 32];
  __shared__ alignas(16) __bf16 sB[2][128 * 32];
  int nbn = N >> 7;
  int bm = blockIdx.x / nbn, bn = blockIdx.x % nbn;
  int tid = threadIdx.x;
  int wv = tid >> 5, lane = tid & 31;
  int wm = wv >> 1, wn = wv & 1;
  int r = lane & 15, hi = lane >> 4;

  const __bf16* Ab = A + (size_t)(bm * 128) * K;
  const __bf16* Bb = Bw + (size_t)(bn * 128) * K;

  auto stage = [&](const __bf16* gbase, __bf16* lbase, int k0) {
#pragma unroll
    for (int rep = 0; rep < 2; rep++) {
      int ch = tid + rep * 256;      // 512 16B-chunks per 128x32 tile
      int row = ch >> 2, c = ch & 3;
      const __bf16* g = gbase + (size_t)row * K + k0 + c * 8;
      __bf16* l = lbase + row * 32 + ((c ^ ((row >> 2) & 3)) << 3);
      cp16(g, l);
    }
  };

  v8f acc[2][4];
#pragma unroll
  for (int i = 0; i < 2; i++)
#pragma unroll
    for (int j = 0; j < 4; j++) acc[i][j] = vzero8();

  int nk = K >> 5;
  stage(Ab, &sA[0][0], 0);
  stage(Bb, &sB[0][0], 0);
  await_async();
  __syncthreads();
  for (int kk = 0; kk < nk; kk++) {
    int p = kk & 1;
    if (kk + 1 < nk) {
      stage(Ab, &sA[p ^ 1][0], (kk + 1) * 32);
      stage(Bb, &sB[p ^ 1][0], (kk + 1) * 32);
    }
    bf16x16 a0 = lds_frag(&sA[p][0], wm * 32 + r, hi);
    bf16x16 a1 = lds_frag(&sA[p][0], wm * 32 + 16 + r, hi);
#pragma unroll
    for (int j = 0; j < 4; j++) {
      bf16x16 bfr = lds_frag(&sB[p][0], wn * 64 + j * 16 + r, hi);
      acc[0][j] = wmma_bf16f32(a0, bfr, acc[0][j]);
      acc[1][j] = wmma_bf16f32(a1, bfr, acc[1][j]);
    }
    await_async();
    __syncthreads();
  }

#pragma unroll
  for (int i = 0; i < 2; i++)
#pragma unroll
    for (int j = 0; j < 4; j++) {
      int grow = bm * 128 + wm * 32 + i * 16 + hi * 8;
      int gcol = bn * 128 + wn * 64 + j * 16 + r;
#pragma unroll
      for (int v = 0; v < 8; v++) {
        size_t idx = (size_t)(grow + v) * N + gcol;
        float o = acc[i][j][v];
        if (resid) o += resid[idx];
        C[idx] = o;
      }
    }
}

// ---------------- RoPE + split into Q/K/V buffers ----------------
// qkv: [4096][2048] f32.  q -> [B][NQ][S][64] bf16 (pre-scaled by 1/8),
// k -> [B][NKV][S][64] bf16, v -> transposed [B][NKV][64][S] bf16.
__global__ void rope_split_kernel(const float* __restrict__ qkv,
                                  const int* __restrict__ start_pos,
                                  __bf16* __restrict__ qb,
                                  __bf16* __restrict__ kb,
                                  __bf16* __restrict__ vT) {
  int bs = blockIdx.x; // 0..4095
  int b = bs >> 11, s = bs & 2047;
  float pos = (float)(start_pos[0] + s);
  const float* row = qkv + (size_t)bs * QKV_OUT;
  const float kLn1e4Over32 = 0.28782313662425575f; // ln(10000)/32
  for (int p = threadIdx.x; p < 1024; p += blockDim.x) {
    float e0 = row[2 * p], e1 = row[2 * p + 1];
    if (p < 768) { // q
      int h = p >> 5, dh = p & 31;
      float ang = pos * __expf(-(float)dh * kLn1e4Over32);
      float sn, cs;
      __sincosf(ang, &sn, &cs);
      __bf16* q = qb + ((size_t)(b * NQ + h) * S_ + s) * HEAD_DIM + 2 * dh;
      q[0] = (__bf16)((e0 * cs - e1 * sn) * 0.125f);
      q[1] = (__bf16)((e1 * cs + e0 * sn) * 0.125f);
    } else if (p < 896) { // k
      int pk = p - 768;
      int h = pk >> 5, dh = pk & 31;
      float ang = pos * __expf(-(float)dh * kLn1e4Over32);
      float sn, cs;
      __sincosf(ang, &sn, &cs);
      __bf16* k = kb + ((size_t)(b * NKV + h) * S_ + s) * HEAD_DIM + 2 * dh;
      k[0] = (__bf16)(e0 * cs - e1 * sn);
      k[1] = (__bf16)(e1 * cs + e0 * sn);
    } else { // v, stored transposed [d][s]
      int pv = p - 896; // 0..127
      int h = pv >> 5;
      int d = (2 * pv) & 63;
      __bf16* vt = vT + (size_t)(b * NKV + h) * HEAD_DIM * S_;
      vt[(size_t)d * S_ + s] = (__bf16)e0;
      vt[(size_t)(d + 1) * S_ + s] = (__bf16)e1;
    }
  }
}

// ---------------- Flash attention, sliding window, one wave per 16 queries ---
// S^T = K · Q^T  (keys in M/VGPRs, queries in lanes) so softmax reduction is
// 16 in-lane values + one shfl_xor(16). P^T's C-layout coincides with the
// B-fragment layout needed for O^T = V^T · P^T -- no data movement.
__global__ __launch_bounds__(128) void attn_kernel(
    const __bf16* __restrict__ q, const __bf16* __restrict__ k,
    const __bf16* __restrict__ vT, const int* __restrict__ start_pos,
    __bf16* __restrict__ out) {
  int wave = (int)(blockIdx.x * (blockDim.x >> 5) + (threadIdx.x >> 5));
  int qtile = wave & (S_ / 16 - 1);
  int bh = wave >> 7; // S_/16 == 128
  int h = bh % NQ, b = bh / NQ;
  int kvh = h / (NQ / NKV);
  int lane = threadIdx.x & 31;
  int r = lane & 15, hi = lane >> 4;
  int q0 = qtile * 16;
  int sp = start_pos[0];

  const __bf16* qbase = q + (size_t)(b * NQ + h) * S_ * HEAD_DIM;
  const __bf16* kbase = k + (size_t)(b * NKV + kvh) * S_ * HEAD_DIM;
  const __bf16* vbase = vT + (size_t)(b * NKV + kvh) * HEAD_DIM * S_;

  const __bf16* qrow = qbase + (size_t)(q0 + r) * HEAD_DIM;
  bf16x16 qf0 = load_frag(qrow, hi);      // d 0..31
  bf16x16 qf1 = load_frag(qrow + 32, hi); // d 32..63

  v8f o[4];
#pragma unroll
  for (int t = 0; t < 4; t++) o[t] = vzero8();
  float m_run = -3.0e38f, l_run = 0.f;
  int i = sp + q0 + r; // absolute query position (this lane's query)

  int kstart = q0 - WINDOW;
  if (kstart < 0) kstart = 0;
  kstart &= ~31;
  for (int kt = kstart; kt <= q0 + 15; kt += 32) {
    v8f s0 = vzero8(), s1 = vzero8();
    {
      const __bf16* krow = kbase + (size_t)(kt + r) * HEAD_DIM;
      s0 = wmma_bf16f32(load_frag(krow, hi), qf0, s0);
      s0 = wmma_bf16f32(load_frag(krow + 32, hi), qf1, s0);
      const __bf16* krow2 = kbase + (size_t)(kt + 16 + r) * HEAD_DIM;
      s1 = wmma_bf16f32(load_frag(krow2, hi), qf0, s1);
      s1 = wmma_bf16f32(load_frag(krow2 + 32, hi), qf1, s1);
    }
    // mask + online softmax (lane holds query i, keys kt+hi*8+v and +16)
    float sv[16];
    float mx = -3.0e38f;
#pragma unroll
    for (int v = 0; v < 8; v++) {
      int j0 = kt + hi * 8 + v;
      int j1 = j0 + 16;
      float a = (j0 <= i && j0 >= i - WINDOW) ? s0[v] : -3.0e38f;
      float bb = (j1 <= i && j1 >= i - WINDOW) ? s1[v] : -3.0e38f;
      sv[v] = a;
      sv[8 + v] = bb;
      mx = fmaxf(mx, fmaxf(a, bb));
    }
    mx = fmaxf(mx, __shfl_xor(mx, 16, 32));
    float m_new = fmaxf(m_run, mx);
    float corr = __expf(m_run - m_new);
    float rs = 0.f;
    bf16x16 pf;
#pragma unroll
    for (int v = 0; v < 8; v++) {
      float e0 = __expf(sv[v] - m_new);
      float e1 = __expf(sv[8 + v] - m_new);
      rs += e0 + e1;
      pf[v] = (__bf16)e0;     // B-frag element v  <-> key kt+hi*8+v
      pf[8 + v] = (__bf16)e1; // element 8+v       <-> key kt+16+hi*8+v
    }
    rs += __shfl_xor(rs, 16, 32);
    l_run = l_run * corr + rs;
    m_run = m_new;
#pragma unroll
    for (int t = 0; t < 4; t++) {
      o[t] = o[t] * corr;
      o[t] = wmma_bf16f32(load_frag(vbase + (size_t)(t * 16 + r) * S_ + kt, hi),
                          pf, o[t]);
    }
  }
  float invl = 1.0f / l_run;
  // O^T layout: lane's query = r; element (t,v) is head-dim d = 16t + 8*hi + v
  __bf16* orow = out + ((size_t)(b * S_) + q0 + r) * HIDDEN + h * HEAD_DIM;
#pragma unroll
  for (int t = 0; t < 4; t++)
#pragma unroll
    for (int v = 0; v < 8; v++)
      orow[t * 16 + hi * 8 + v] = (__bf16)(o[t][v] * invl);
}

extern "C" void kernel_launch(void* const* d_in, const int* in_sizes, int n_in,
                              void* d_out, int out_size, void* d_ws,
                              size_t ws_size, hipStream_t stream) {
  (void)in_sizes; (void)n_in; (void)out_size; (void)ws_size;
  const float* x = (const float*)d_in[0];
  const float* norm_scale = (const float*)d_in[1];
  const float* w_qkv = (const float*)d_in[2];
  const float* w_out = (const float*)d_in[3];
  const int* start_pos = (const int*)d_in[4];

  char* ws = (char*)d_ws;
  auto carve = [&](size_t bytes) -> char* {
    char* p = ws;
    ws += (bytes + 255) & ~(size_t)255;
    return p;
  };
  __bf16* xn = (__bf16*)carve((size_t)ROWS * HIDDEN * 2);
  __bf16* wqkv = (__bf16*)carve((size_t)QKV_OUT * HIDDEN * 2);
  __bf16* wout = (__bf16*)carve((size_t)HIDDEN * HIDDEN * 2);
  float* qkv = (float*)carve((size_t)ROWS * QKV_OUT * 4);
  __bf16* qb = (__bf16*)carve((size_t)B_ * NQ * S_ * HEAD_DIM * 2);
  __bf16* kb = (__bf16*)carve((size_t)B_ * NKV * S_ * HEAD_DIM * 2);
  __bf16* vT = (__bf16*)carve((size_t)B_ * NKV * HEAD_DIM * S_ * 2);
  __bf16* attn = (__bf16*)carve((size_t)ROWS * HIDDEN * 2);

  cvt_bf16_kernel<<<2048, 256, 0, stream>>>(w_qkv, wqkv, QKV_OUT * HIDDEN);
  cvt_bf16_kernel<<<2048, 256, 0, stream>>>(w_out, wout, HIDDEN * HIDDEN);
  rmsnorm_kernel<<<ROWS, 256, 0, stream>>>(x, norm_scale, xn);
  // qkv = xn @ wqkv^T : M=4096 N=2048 K=1536 -> (32 x 16) 128x128 tiles
  gemm_bf16_lds_kernel<<<(ROWS / 128) * (QKV_OUT / 128), 256, 0, stream>>>(
      xn, wqkv, nullptr, qkv, ROWS, QKV_OUT, HIDDEN);
  rope_split_kernel<<<ROWS, 256, 0, stream>>>(qkv, start_pos, qb, kb, vT);
  // 2*24*128 = 6144 waves, 4 waves/block
  attn_kernel<<<(B_ * NQ * (S_ / 16)) / 4, 128, 0, stream>>>(qb, kb, vT,
                                                             start_pos, attn);
  // out = x + attn @ wout^T : M=4096 N=1536 K=1536 -> (32 x 12) tiles
  gemm_bf16_lds_kernel<<<(ROWS / 128) * (HIDDEN / 128), 256, 0, stream>>>(
      attn, wout, x, (float*)d_out, ROWS, HIDDEN, HIDDEN);
}
